// ConvLinformerSelfAttention_49082886259386
// MI455X (gfx1250) — matrix operand order repaired and checked by reading, not compiled, not run on hardware
//
#include <hip/hip_runtime.h>
#include <hip/hip_bf16.h>
#include <math.h>

// ---------------------------------------------------------------------------
// CDNA5 (gfx1250) wave32 WMMA types
// ---------------------------------------------------------------------------
typedef __attribute__((ext_vector_type(16))) _Float16 v16h;
typedef __attribute__((ext_vector_type(8)))  _Float16 v8h;
typedef __attribute__((ext_vector_type(8)))  float    v8f;
typedef __attribute__((ext_vector_type(4)))  unsigned v4u;
typedef __attribute__((ext_vector_type(4)))  int      v4i;
typedef __attribute__((ext_vector_type(8)))  int      v8i;

#define WMMA_F16(a, b, c) \
  __builtin_amdgcn_wmma_f32_16x16x32_f16(false, (a), false, (b), (short)0, (c), false, false)

// Problem constants
#define Bz   4
#define Nn   4096
#define Dd   1024
#define Hh   8
#define DH   128
#define Kk   256
#define KER  32

#if __has_builtin(__builtin_amdgcn_tensor_load_to_lds)
#define HAVE_TDM 1
#endif
#if __has_builtin(__builtin_amdgcn_s_wait_tensorcnt)
#define WAIT_TENSORCNT0() __builtin_amdgcn_s_wait_tensorcnt(0)
#else
#define WAIT_TENSORCNT0()
#endif

// ---------------------------------------------------------------------------
// Tensor Data Mover: 2D tile (rows x row_elems of f16, row stride in elems)
// from global into LDS, per D# layout in CDNA5 ISA.
// Group0: count=1 | lds_addr | global_addr | type=2.  Group1: data_size=2B,
// tensor_dim0/1 (OOB), tile_dim0/1, tensor_dim0_stride.  Groups 2/3 zero (2D).
// ---------------------------------------------------------------------------
#ifdef HAVE_TDM
__device__ __forceinline__ void tdm_load_2d_f16(unsigned lds_addr, unsigned long long gaddr,
                                                unsigned rows, unsigned row_elems,
                                                unsigned row_stride_elems,
                                                unsigned tensor_d0, unsigned tensor_d1) {
  v4u g0;
  g0[0] = 1u;                                               // count=1 (user D#)
  g0[1] = lds_addr;                                         // bits 63:32
  g0[2] = (unsigned)(gaddr & 0xffffffffull);                // bits 95:64
  g0[3] = (unsigned)((gaddr >> 32) & 0x01ffffffull) | (2u << 30);  // addr[56:32] | type=2
  v8i g1;
  g1[0] = (int)(1u << 16);                                  // data_size=1 (2 bytes)
  g1[1] = (int)((tensor_d0 & 0xffffu) << 16);               // tensor_dim0[15:0] @ bits 63:48
  g1[2] = (int)((tensor_d0 >> 16) | ((tensor_d1 & 0xffffu) << 16)); // d0[31:16] | d1[15:0]
  g1[3] = (int)((tensor_d1 >> 16) | (row_elems << 16));     // d1[31:16] | tile_dim0
  g1[4] = (int)(rows & 0xffffu);                            // tile_dim1 (tile_dim2=0)
  g1[5] = (int)row_stride_elems;                            // tensor_dim0_stride[31:0]
  g1[6] = 0;                                                // stride[47:32] | dim1_stride lo
  g1[7] = 0;
  v4i z4 = {0, 0, 0, 0};
#if __clang_major__ >= 23
  v8i z8 = {0, 0, 0, 0, 0, 0, 0, 0};
  __builtin_amdgcn_tensor_load_to_lds(g0, g1, z4, z4, z8, 0);
#else
  __builtin_amdgcn_tensor_load_to_lds(g0, g1, z4, z4, 0);
#endif
}
#endif

// ---------------------------------------------------------------------------
// Fragment loaders (16x16x32 f16 WMMA, wave32; k = 16*(lane/16) + s in A and B)
// ---------------------------------------------------------------------------
__device__ __forceinline__ v16h load_a_frag(const _Float16* base, int row_stride, int lane) {
  const int m  = lane & 15;
  const int hi = lane >> 4;
  const _Float16* p = base + m * row_stride + hi * 16;
  v8h lo = *(const v8h*)(p);
  v8h hh = *(const v8h*)(p + 8);
  v16h r;
#pragma unroll
  for (int i = 0; i < 8; ++i) { r[i] = lo[i]; r[i + 8] = hh[i]; }
  return r;
}

__device__ __forceinline__ v16h load_b_frag(const _Float16* bt_base, int row_stride, int lane) {
  const int n  = lane & 15;
  const int hi = lane >> 4;
  return *(const v16h*)(bt_base + n * row_stride + hi * 16);
}

// ---------------------------------------------------------------------------
// fp32 -> fp16 conversion / packing kernels
// ---------------------------------------------------------------------------
__global__ void cvt_f32_f16(const float* __restrict__ src, _Float16* __restrict__ dst, int n) {
  int i = blockIdx.x * 256 + threadIdx.x;
  if (i < n) dst[i] = (_Float16)src[i];
}

// Wpk [K][N][KER] fp32  ->  [KER][K][N] fp16 (contiguous n per tap for A-frags)
__global__ void pack_wpk(const float* __restrict__ src, _Float16* __restrict__ dst) {
  int i = blockIdx.x * 256 + threadIdx.x;           // 32*256*4096 = 2^25
  if (i >= KER * Kk * Nn) return;
  int n = i & (Nn - 1);
  int k = (i >> 12) & (Kk - 1);
  int t = i >> 20;
  dst[i] = (_Float16)src[((size_t)(k * Nn + n)) * KER + t];
}

// keysK [B][K][D] fp16 -> vT [(b*H+h)*DH + d][K] fp16 (k contiguous for B-frags)
__global__ void pack_vT(const _Float16* __restrict__ keysK, _Float16* __restrict__ vT) {
  int i = blockIdx.x * 256 + threadIdx.x;           // 4*8*128*256 = 2^20
  if (i >= Bz * Hh * DH * Kk) return;
  int kk = i & (Kk - 1);
  int d  = (i >> 8) & (DH - 1);
  int h  = (i >> 15) & (Hh - 1);
  int b  = i >> 18;
  vT[i] = keysK[((size_t)(b * Kk + kk)) * Dd + h * DH + d];
}

// ---------------------------------------------------------------------------
// Generic WMMA GEMM:  C(MxN) = A(MxKdim) @ Bt(NxKdim)^T  [+ bias]
// Block: 256 threads = 8 waves. Tile: 128(M) x 64(N).
// Bt chunk staged into LDS by the Tensor Data Mover (TDM) when available.
// ---------------------------------------------------------------------------
template <bool OUT_F16, bool HAS_BIAS>
__global__ void gemm_bt(const _Float16* __restrict__ A, const _Float16* __restrict__ Bt,
                        const float* __restrict__ bias, void* __restrict__ Cout,
                        int M, int N, int Kdim) {
  __shared__ __attribute__((aligned(32))) _Float16 bts[64 * 32];
  const int tid = threadIdx.x, lane = tid & 31, wv = tid >> 5;
  const int row0 = blockIdx.y * 128 + wv * 16;
  const int col0 = blockIdx.x * 64;
#ifdef HAVE_TDM
  const unsigned bts_lds = (unsigned)(size_t)&bts[0];   // LDS byte offset (low 32b of flat)
#endif

  v8f acc[4] = {};
  for (int kc = 0; kc < Kdim; kc += 32) {
#ifdef HAVE_TDM
    if (wv == 0) {  // one wave drives the TDM; tracked by TENSORcnt
      tdm_load_2d_f16(bts_lds,
                      (unsigned long long)(Bt + (size_t)col0 * Kdim + kc),
                      /*rows=*/64, /*row_elems=*/32, /*row_stride=*/(unsigned)Kdim,
                      /*tensor_d0=*/(unsigned)Kdim, /*tensor_d1=*/(unsigned)N);
    }
#else
    {  // cooperative stage of Bt[col0..col0+63][kc..kc+31] into LDS
      int idx = tid * 8;
      int r = idx >> 5, c = idx & 31;
      *(v8h*)(bts + idx) = *(const v8h*)(Bt + (size_t)(col0 + r) * Kdim + kc + c);
    }
#endif
    // Issue A-fragment loads before waiting on the B tile (overlap latency),
    // and prefetch the next K-chunk of A into cache.
    v16h a = load_a_frag(A + (size_t)row0 * Kdim + kc, Kdim, lane);
    if (kc + 32 < Kdim)
      __builtin_prefetch(A + (size_t)row0 * Kdim + kc + 32 + (lane << 3), 0, 1);
#ifdef HAVE_TDM
    if (wv == 0) WAIT_TENSORCNT0();
#endif
    __syncthreads();
#pragma unroll
    for (int jt = 0; jt < 4; ++jt) {
      v16h b = load_b_frag(bts + jt * 16 * 32, 32, lane);
      acc[jt] = WMMA_F16(a, b, acc[jt]);
    }
    __syncthreads();
  }

  const int mbase = (lane < 16) ? 0 : 8;
  const int cl = lane & 15;
#pragma unroll
  for (int jt = 0; jt < 4; ++jt) {
#pragma unroll
    for (int r = 0; r < 8; ++r) {
      int row = row0 + mbase + r;
      int col = col0 + jt * 16 + cl;
      float v = acc[jt][r];
      if (HAS_BIAS) v += bias[col];
      if (OUT_F16) ((_Float16*)Cout)[(size_t)row * N + col] = (_Float16)v;
      else         ((float*)Cout)[(size_t)row * N + col] = v;
    }
  }
}

// ---------------------------------------------------------------------------
// Conv1d as tap-shifted GEMM:
//   keysK[b,k,d] = sum_t sum_n Wpk[k,n,t] * keys[b,n,d+t-15] + bpk[k]
// Keys tile (32 n x 96 d incl. halo) staged *transposed* in LDS so every
// tap's B-fragment is one contiguous 32B read.  Grid: (D/64, 256/128, B).
// ---------------------------------------------------------------------------
__global__ void conv_gemm(const _Float16* __restrict__ Wp, const _Float16* __restrict__ keys,
                          const float* __restrict__ bpk, _Float16* __restrict__ keysK) {
  __shared__ __attribute__((aligned(32))) _Float16 kts[96 * 32];  // [d_local][n_local]
  const int tid = threadIdx.x, lane = tid & 31, wv = tid >> 5;
  const int b  = blockIdx.z;
  const int k0 = blockIdx.y * 128 + wv * 16;
  const int d0 = blockIdx.x * 64;

  v8f acc[4] = {};
  for (int nc = 0; nc < Nn; nc += 32) {
    for (int idx = tid; idx < 96 * 32; idx += 256) {
      int nl = idx / 96, dl = idx % 96;
      int d = d0 + dl - 15;                       // 'same' padding (15,16)
      _Float16 v = (_Float16)0.f;
      if (d >= 0 && d < Dd) v = keys[((size_t)(b * Nn + nc + nl)) * Dd + d];
      kts[dl * 32 + nl] = v;
    }
    if (nc + 32 < Nn)
      __builtin_prefetch(keys + ((size_t)(b * Nn + nc + 32 + (tid >> 3))) * Dd + d0, 0, 1);
    __syncthreads();
#pragma unroll 4
    for (int t = 0; t < KER; ++t) {
      v16h a = load_a_frag(Wp + ((size_t)(t * Kk + k0)) * Nn + nc, Nn, lane);
#pragma unroll
      for (int jt = 0; jt < 4; ++jt) {
        v16h bb = load_b_frag(kts + (jt * 16 + t) * 32, 32, lane);
        acc[jt] = WMMA_F16(a, bb, acc[jt]);
      }
    }
    __syncthreads();
  }

  const int mbase = (lane < 16) ? 0 : 8;
  const int cl = lane & 15;
#pragma unroll
  for (int jt = 0; jt < 4; ++jt) {
#pragma unroll
    for (int r = 0; r < 8; ++r) {
      int k = k0 + mbase + r;
      int d = d0 + jt * 16 + cl;
      keysK[((size_t)(b * Kk + k)) * Dd + d] = (_Float16)(acc[jt][r] + bpk[k]);
    }
  }
}

// ---------------------------------------------------------------------------
// Linformer attention for one (b,h): dots = q k^T * dh^-0.5 ; softmax ; @ v.
// Block: 64 threads = 2 waves; each wave owns 16 rows of n.
// 1/rowsum folded into the second GEMM's epilogue (unnormalized exp in LDS).
// ---------------------------------------------------------------------------
__global__ void linformer_attn(const _Float16* __restrict__ q, const _Float16* __restrict__ kv,
                               const _Float16* __restrict__ vT, _Float16* __restrict__ outp) {
  __shared__ __attribute__((aligned(32))) float    dots[2][16][256];
  __shared__ __attribute__((aligned(32))) _Float16 attn[2][16][256];
  __shared__ float red[2][16][2];
  __shared__ float invs[2][16];

  const int tid = threadIdx.x, lane = tid & 31, wv = tid >> 5;
  const int bh = blockIdx.x;
  const int b = bh >> 3, h = bh & 7;
  const int n0 = blockIdx.y * 32 + wv * 16;
  const float scale = 0.08838834764831845f;  // DH^-0.5

  v16h qf[4];
#pragma unroll
  for (int kc = 0; kc < 4; ++kc)
    qf[kc] = load_a_frag(q + ((size_t)(b * Nn + n0)) * Dd + h * DH + kc * 32, Dd, lane);

  const int mbase = (lane < 16) ? 0 : 8;
  const int cl = lane & 15;

  // dots(16x256) = q(16x128) @ k^T : 64 WMMAs
  for (int jt = 0; jt < 16; ++jt) {
    v8f acc = {};
#pragma unroll
    for (int kc = 0; kc < 4; ++kc) {
      v16h bb = load_b_frag(kv + ((size_t)(b * Kk + jt * 16)) * Dd + h * DH + kc * 32, Dd, lane);
      acc = WMMA_F16(qf[kc], bb, acc);
    }
#pragma unroll
    for (int r = 0; r < 8; ++r)
      dots[wv][mbase + r][jt * 16 + cl] = acc[r] * scale;
  }
  __syncthreads();

  // softmax bookkeeping: 2 lanes per row, 128 cols each
  const int m = lane & 15, hh = lane >> 4;
  float mx = -1e30f;
  for (int c = 0; c < 128; ++c) mx = fmaxf(mx, dots[wv][m][hh * 128 + c]);
  red[wv][m][hh] = mx;
  __syncthreads();
  float rmax = fmaxf(red[wv][m][0], red[wv][m][1]);
  float s = 0.f;
  for (int c = 0; c < 128; ++c) {
    float e = __expf(dots[wv][m][hh * 128 + c] - rmax);
    attn[wv][m][hh * 128 + c] = (_Float16)e;
    s += e;
  }
  red[wv][m][hh] = s;
  __syncthreads();
  if (hh == 0) invs[wv][m] = 1.f / (red[wv][m][0] + red[wv][m][1]);
  __syncthreads();

  // out(16x128) = attn(16x256) @ v(256x128) : 64 WMMAs  (v from transposed vT)
  v8f acc2[8] = {};
  for (int kc = 0; kc < 8; ++kc) {
    v16h a = load_a_frag(&attn[wv][0][0] + kc * 32, 256, lane);
#pragma unroll
    for (int jt = 0; jt < 8; ++jt) {
      v16h bb = load_b_frag(vT + (((size_t)(b * Hh + h)) * DH + jt * 16) * Kk + kc * 32, Kk, lane);
      acc2[jt] = WMMA_F16(a, bb, acc2[jt]);
    }
  }
#pragma unroll
  for (int jt = 0; jt < 8; ++jt) {
#pragma unroll
    for (int r = 0; r < 8; ++r) {
      int row = mbase + r;
      float o = acc2[jt][r] * invs[wv][row];
      outp[((size_t)(b * Nn + n0 + row)) * Dd + h * DH + jt * 16 + cl] = (_Float16)o;
    }
  }
}

// ---------------------------------------------------------------------------
// Host-side orchestration (graph-capture safe: kernel launches only)
// ---------------------------------------------------------------------------
extern "C" void kernel_launch(void* const* d_in, const int* in_sizes, int n_in,
                              void* d_out, int out_size, void* d_ws, size_t ws_size,
                              hipStream_t stream) {
  const float* x   = (const float*)d_in[0];
  const float* Wq  = (const float*)d_in[1];
  const float* Wk  = (const float*)d_in[2];
  const float* Wpk = (const float*)d_in[3];
  const float* bpk = (const float*)d_in[4];
  const float* Wo  = (const float*)d_in[5];
  const float* bo  = (const float*)d_in[6];
  float* out = (float*)d_out;

  char* ws = (char*)d_ws;
  _Float16* x_h    = (_Float16*)(ws);                      //  33,554,432 B
  _Float16* wq_h   = (_Float16*)(ws + 33554432ull);        //   2,097,152 B  (Bt for q)
  _Float16* wk_h   = (_Float16*)(ws + 35651584ull);        //   2,097,152 B
  _Float16* wo_h   = (_Float16*)(ws + 37748736ull);        //   2,097,152 B
  _Float16* q_h    = (_Float16*)(ws + 39845888ull);        //  33,554,432 B
  _Float16* keys_h = (_Float16*)(ws + 73400320ull);        //  33,554,432 B
  _Float16* wpk_p  = (_Float16*)(ws + 106954752ull);       //  67,108,864 B
  _Float16* keysK  = (_Float16*)(ws + 174063616ull);       //   2,097,152 B
  _Float16* vT_h   = (_Float16*)(ws + 176160768ull);       //   2,097,152 B
  _Float16* att_h  = (_Float16*)(ws + 178257920ull);       //  33,554,432 B

  // 1) cast inputs to f16 (Bt for the linear layers = the torch weight itself)
  cvt_f32_f16<<<(Bz * Nn * Dd) / 256, 256, 0, stream>>>(x, x_h, Bz * Nn * Dd);
  cvt_f32_f16<<<(Dd * Dd) / 256, 256, 0, stream>>>(Wq, wq_h, Dd * Dd);
  cvt_f32_f16<<<(Dd * Dd) / 256, 256, 0, stream>>>(Wk, wk_h, Dd * Dd);
  cvt_f32_f16<<<(Dd * Dd) / 256, 256, 0, stream>>>(Wo, wo_h, Dd * Dd);
  pack_wpk<<<(KER * Kk * Nn) / 256, 256, 0, stream>>>(Wpk, wpk_p);

  // 2) q = x @ Wq^T, keys = x @ Wk^T   (f16 outputs)
  dim3 gproj(Dd / 64, (Bz * Nn) / 128);
  gemm_bt<true, false><<<gproj, 256, 0, stream>>>(x_h, wq_h, nullptr, q_h, Bz * Nn, Dd, Dd);
  gemm_bt<true, false><<<gproj, 256, 0, stream>>>(x_h, wk_h, nullptr, keys_h, Bz * Nn, Dd, Dd);

  // 3) conv1d over feature axis -> keysK (b, K, D), share_kv values == keys
  dim3 gconv(Dd / 64, Kk / 128, Bz);
  conv_gemm<<<gconv, 256, 0, stream>>>(wpk_p, keys_h, bpk, keysK);
  pack_vT<<<(Bz * Hh * DH * Kk) / 256, 256, 0, stream>>>(keysK, vT_h);

  // 4) attention per (b,h)
  dim3 gattn(Bz * Hh, Nn / 32);
  linformer_attn<<<gattn, 64, 0, stream>>>(q_h, keysK, vT_h, att_h);

  // 5) out = att @ Wo^T + bo  (f32 output)
  gemm_bt<false, true><<<gproj, 256, 0, stream>>>(att_h, wo_h, bo, out, Bz * Nn, Dd, Dd);
}
